// pos_radar_DynamicEmbedder_90950227460832
// MI455X (gfx1250) — matrix operand order repaired and checked by reading, not compiled, not run on hardware
//
#include <hip/hip_runtime.h>

typedef __attribute__((ext_vector_type(16))) _Float16 v16h;
typedef __attribute__((ext_vector_type(8)))  _Float16 v8h;
typedef __attribute__((ext_vector_type(8)))  float    v8f;
typedef __attribute__((ext_vector_type(4)))  float    v4f;

#define GH 512
#define GW 512
#define GHW (GH*GW)
#define WPB 4   // waves per block (wave32)

// SHIFTS order must match reference: [0,0],[-1,0],[1,0],[0,1],[-1,1],[1,1],[0,-1],[-1,-1],[1,-1]
__device__ __constant__ int c_dr[9] = {0,-1,1,0,-1,1,0,-1,1};
__device__ __constant__ int c_dc[9] = {0,0,0,1,1,1,-1,-1,-1};

__global__ void fill_grid_k(int* __restrict__ g) {
    int i = blockIdx.x * blockDim.x + threadIdx.x;
    if (i < GHW) g[i] = -1;
}

__global__ void scatter_grid_k(const int* __restrict__ coors, int* __restrict__ g, int n) {
    int i = blockIdx.x * blockDim.x + threadIdx.x;
    if (i < n) g[coors[2*i] * GW + coors[2*i+1]] = i;
}

__global__ void zero_canvas_k(v4f* __restrict__ p, int n4) {
    int i = blockIdx.x * blockDim.x + threadIdx.x;
    if (i < n4) {
        v4f z = {0.f, 0.f, 0.f, 0.f};
        __builtin_nontemporal_store(z, p + i);
    }
}

__device__ inline v16h cat16(v8h lo, v8h hi) {
    return __builtin_shufflevector(lo, hi, 0,1,2,3,4,5,6,7,8,9,10,11,12,13,14,15);
}

// A fragment (16x32 f16, M x K): lane l (m=l&15, hf=l>>4):
//   halves 0..7  = row m, K = hf*8 .. +7 ; halves 8..15 = row m, K = 16+hf*8 .. +7
__device__ inline v16h load_a_frag(const _Float16* row, int hf) {
    v8h lo = *(const v8h*)(row + hf*8);
    v8h hi = *(const v8h*)(row + 16 + hf*8);
    return cat16(lo, hi);
}

// B fragment (32x16 f16, K x N), B(k,n) = W[n][k]: lane l (n=l&15, hf=l>>4) holds
// col n, halves 0..7 = K hf*8 .. +7 ; halves 8..15 = K 16+hf*8 .. +7 (matches A).
__device__ inline v16h load_b_frag(const _Float16* wrow, int hf) {
    v8h lo = *(const v8h*)(wrow + hf*8);
    v8h hi = *(const v8h*)(wrow + 16 + hf*8);
    return cat16(lo, hi);
}

__device__ inline v8f wmma16(v16h a, v16h b, v8f c) {
    return __builtin_amdgcn_wmma_f32_16x16x32_f16(false, a, false, b, (short)0, c, false, false);
}

__global__ __launch_bounds__(128) void attn_kernel(
    const float* __restrict__ feats, const int* __restrict__ coors,
    const float* __restrict__ pos,   const float* __restrict__ wqkv,
    const float* __restrict__ bqkv,  const float* __restrict__ wout,
    const float* __restrict__ bout,  const int* __restrict__ grid,
    float* __restrict__ canvas, int n)
{
    __shared__ _Float16 sW[96][32];        // (0.25*wq)|wk|wv rows, f16
    __shared__ _Float16 sWo[32][32];       // w_out, f16
    __shared__ float    sPosF[9][32];
    __shared__ float    sB[96];            // (0.25*bq)|bk|bv
    __shared__ float    sBo[32];
    __shared__ int      sRC[WPB][16][2];
    __shared__ _Float16 sKV[WPB][16][32];  // f16 A-tile staging
    __shared__ float    sQ[WPB][16][33];   // padded to dodge bank conflicts
    __shared__ float    sK[WPB][16][33];
    __shared__ float    sV[WPB][16][33];
    __shared__ _Float16 sAO[WPB][16][32];

    const int tid = threadIdx.x;
    // fold softmax scale 1/sqrt(DH)=0.25 into Wq/bq at staging time
    for (int idx = tid; idx < 96*32; idx += 128) {
        float v = wqkv[idx];
        sW[idx>>5][idx&31] = (_Float16)(idx < 32*32 ? v * 0.25f : v);
    }
    for (int idx = tid; idx < 32*32; idx += 128) sWo[idx>>5][idx&31] = (_Float16)wout[idx];
    for (int idx = tid; idx < 9*32;  idx += 128) sPosF[idx>>5][idx&31] = pos[idx];
    if (tid < 96) sB[tid]  = (tid < 32 ? bqkv[tid] * 0.25f : bqkv[tid]);
    if (tid < 32) sBo[tid] = bout[tid];

    const int w    = tid >> 5;
    const int lane = tid & 31;
    const int m    = lane & 15;
    const int hf   = lane >> 4;
    const int base = (blockIdx.x * WPB + w) * 16;
    const int i    = base + m;
    const bool act = (i < n);
    const int ic   = act ? i : (n - 1);
    const int rr   = coors[2*ic], cc = coors[2*ic+1];
    if (hf == 0) { sRC[w][m][0] = rr; sRC[w][m][1] = cc; }

    // hoisted neighbor-index gather: 9 independent scattered loads, full MLP
    int gidx[9];
    #pragma unroll
    for (int j = 0; j < 9; ++j) {
        int sr = rr + c_dr[j], sc2 = cc + c_dc[j];
        bool ok = (sr >= 0) & (sr < GH) & (sc2 >= 0) & (sc2 < GW);
        gidx[j] = ok ? grid[sr*GW + sc2] : -1;
    }

    // stage this tile's feats as f16 (A-tile for the Q GEMM)
    {
        const v4f* fp = (const v4f*)(feats + (size_t)ic*32 + hf*16);
        v4f f[4]; f[0]=fp[0]; f[1]=fp[1]; f[2]=fp[2]; f[3]=fp[3];
        _Float16* dst = &sKV[w][m][hf*16];
        #pragma unroll
        for (int x = 0; x < 4; ++x) {
            dst[4*x+0]=(_Float16)f[x].x; dst[4*x+1]=(_Float16)f[x].y;
            dst[4*x+2]=(_Float16)f[x].z; dst[4*x+3]=(_Float16)f[x].w;
        }
    }
    __syncthreads();

    // ---- Q = feats @ (0.25*Wq)^T + 0.25*bq via WMMA ----
    {
        v16h a = load_a_frag(&sKV[w][m][0], hf);
        #pragma unroll
        for (int t = 0; t < 2; ++t) {
            v16h b = load_b_frag(&sW[t*16 + m][0], hf);
            v8f c = {};
            c = wmma16(a, b, c);
            float bb = sB[t*16 + m];
            #pragma unroll
            for (int r = 0; r < 8; ++r) sQ[w][r + hf*8][t*16 + m] = c[r] + bb;
        }
    }
    // persistent K/V weight B-fragments
    v16h wkB0 = load_b_frag(&sW[32 + m][0], hf);
    v16h wkB1 = load_b_frag(&sW[48 + m][0], hf);
    v16h wvB0 = load_b_frag(&sW[64 + m][0], hf);
    v16h wvB1 = load_b_frag(&sW[80 + m][0], hf);
    __syncthreads();

    // this lane owns (voxel m, head hf); scale already folded into q
    float qv[16];
    #pragma unroll
    for (int c2 = 0; c2 < 16; ++c2) qv[c2] = sQ[w][m][hf*16 + c2];

    float mrun = -3.4e38f, srun = 0.f, acc[16];
    #pragma unroll
    for (int c2 = 0; c2 < 16; ++c2) acc[c2] = 0.f;

    // prologue of the gather pipeline: fetch shift-0 neighbor row
    v4f pf[4];
    {
        int g = gidx[0];
        if (g >= 0) {
            const v4f* fp = (const v4f*)(feats + (size_t)g*32 + hf*16);
            pf[0]=fp[0]; pf[1]=fp[1]; pf[2]=fp[2]; pf[3]=fp[3];
        } else {
            #pragma unroll
            for (int x = 0; x < 4; ++x) pf[x] = (v4f){0.f,0.f,0.f,0.f};
        }
    }

    // ---- streaming attention over the 9 shifts (gather pipelined 1 ahead) ----
    #pragma unroll
    for (int j = 0; j < 9; ++j) {
        // stage kv tile for shift j from prefetched registers
        {
            _Float16* dst = &sKV[w][m][hf*16];
            if (gidx[j] >= 0) {
                const float* pp = &sPosF[j][hf*16];
                #pragma unroll
                for (int x = 0; x < 4; ++x) {
                    dst[4*x+0]=(_Float16)(pf[x].x + pp[4*x+0]);
                    dst[4*x+1]=(_Float16)(pf[x].y + pp[4*x+1]);
                    dst[4*x+2]=(_Float16)(pf[x].z + pp[4*x+2]);
                    dst[4*x+3]=(_Float16)(pf[x].w + pp[4*x+3]);
                }
            } else {
                #pragma unroll
                for (int x = 0; x < 16; ++x) dst[x] = (_Float16)0.f;
            }
        }
        __syncthreads();

        // issue next shift's gather now; latency hidden behind WMMA + softmax
        if (j < 8) {
            int g = gidx[j+1];
            if (g >= 0) {
                const v4f* fp = (const v4f*)(feats + (size_t)g*32 + hf*16);
                pf[0]=fp[0]; pf[1]=fp[1]; pf[2]=fp[2]; pf[3]=fp[3];
            } else {
                #pragma unroll
                for (int x = 0; x < 4; ++x) pf[x] = (v4f){0.f,0.f,0.f,0.f};
            }
        }

        v16h a = load_a_frag(&sKV[w][m][0], hf);
        {
            v8f ck = {}, cv = {};
            ck = wmma16(a, wkB0, ck);
            cv = wmma16(a, wvB0, cv);
            float bk = sB[32 + m], bv = sB[64 + m];
            #pragma unroll
            for (int r = 0; r < 8; ++r) {
                sK[w][r + hf*8][m] = ck[r] + bk;
                sV[w][r + hf*8][m] = cv[r] + bv;
            }
            v8f ck1 = {}, cv1 = {};
            ck1 = wmma16(a, wkB1, ck1);
            cv1 = wmma16(a, wvB1, cv1);
            bk = sB[48 + m]; bv = sB[80 + m];
            #pragma unroll
            for (int r = 0; r < 8; ++r) {
                sK[w][r + hf*8][16 + m] = ck1[r] + bk;
                sV[w][r + hf*8][16 + m] = cv1[r] + bv;
            }
        }
        __syncthreads();

        float s = 0.f;
        #pragma unroll
        for (int c2 = 0; c2 < 16; ++c2) s += qv[c2] * sK[w][m][hf*16 + c2];
        float mn   = fmaxf(mrun, s);
        float corr = __expf(mrun - mn);
        float e    = __expf(s - mn);
        #pragma unroll
        for (int c2 = 0; c2 < 16; ++c2)
            acc[c2] = acc[c2]*corr + e * sV[w][m][hf*16 + c2];
        srun = srun*corr + e;
        mrun = mn;
    }

    // ---- output projection via WMMA, then scatter to canvas (non-temporal) ----
    {
        float inv = 1.0f / srun;
        _Float16* dsta = &sAO[w][m][hf*16];
        #pragma unroll
        for (int c2 = 0; c2 < 16; ++c2) dsta[c2] = (_Float16)(acc[c2]*inv);
    }
    __syncthreads();
    {
        v16h a = load_a_frag(&sAO[w][m][0], hf);
        #pragma unroll
        for (int t = 0; t < 2; ++t) {
            v16h b = load_b_frag(&sWo[t*16 + m][0], hf);
            v8f c = {};
            c = wmma16(a, b, c);
            float bb = sBo[t*16 + m];
            #pragma unroll
            for (int r = 0; r < 8; ++r) {
                int vm = r + hf*8;
                int vi = base + vm;
                if (vi < n) {
                    int vr = sRC[w][vm][0], vc = sRC[w][vm][1];
                    __builtin_nontemporal_store(
                        c[r] + bb,
                        canvas + (size_t)(32 + t*16 + m)*GHW + vr*GW + vc);
                }
            }
        }
    }
    if (act) {
        #pragma unroll
        for (int c2 = 0; c2 < 16; ++c2) {
            int ch = hf*16 + c2;
            __builtin_nontemporal_store(
                feats[(size_t)i*32 + ch],
                canvas + (size_t)ch*GHW + rr*GW + cc);
        }
    }
}

extern "C" void kernel_launch(void* const* d_in, const int* in_sizes, int n_in,
                              void* d_out, int out_size, void* d_ws, size_t ws_size,
                              hipStream_t stream) {
    const float* feats = (const float*)d_in[0];
    const int*   coors = (const int*)d_in[1];
    const float* pos   = (const float*)d_in[2];
    const float* wqkv  = (const float*)d_in[3];
    const float* bqkv  = (const float*)d_in[4];
    const float* wout  = (const float*)d_in[5];
    const float* bout  = (const float*)d_in[6];
    float* canvas = (float*)d_out;
    int*   grid   = (int*)d_ws;
    const int n = in_sizes[0] / 32;

    fill_grid_k<<<GHW/256, 256, 0, stream>>>(grid);
    scatter_grid_k<<<(n + 255)/256, 256, 0, stream>>>(coors, grid, n);
    zero_canvas_k<<<(out_size/4 + 255)/256, 256, 0, stream>>>((v4f*)d_out, out_size/4);
    attn_kernel<<<(n + 16*WPB - 1)/(16*WPB), 32*WPB, 0, stream>>>(
        feats, coors, pos, wqkv, bqkv, wout, bout, grid, canvas, n);
}
